// CLFBlock_80178449482321
// MI455X (gfx1250) — matrix operand order repaired and checked by reading, not compiled, not run on hardware
//
#include <hip/hip_runtime.h>
#include <hip/hip_bf16.h>

// ---------------------------------------------------------------------------
// SNN block: h = x@W1^T + b1 ; LIF scan ; y = mean_t(s@W2^T) + b2 ; log_softmax
// T=32, B=512, D=1024, C=1000.
// GEMMs: v_wmma_f32_16x16x32_f16; tile staging via Tensor Data Mover
// (tensor_load_to_lds + s_wait_tensorcnt), double-buffered LDS.
// Fallback (USE_TDM=0): per-lane global_load_async_to_lds_b128.
// ---------------------------------------------------------------------------

#define USE_TDM 1

typedef __attribute__((ext_vector_type(16))) _Float16 v16h;
typedef __attribute__((ext_vector_type(8)))  float    v8f;
typedef __attribute__((ext_vector_type(4)))  unsigned u32x4;
typedef __attribute__((ext_vector_type(8)))  int      i32x8;

#define T_STEPS 32
#define BATCH   512
#define DIM     1024
#define NCLS    1000
#define M1      (T_STEPS * BATCH)   // 16384

#define B_TM 64
#define B_TN 128
#define B_TK 32

union FragU { uint4 u[2]; v16h v; };
union H4   { _Float16 h[4]; uint2 u; };
union H2   { _Float16 h[2]; unsigned u; };

// ---- CDNA5 async global->LDS copy (ASYNCcnt) -- fallback path -------------
__device__ __forceinline__ void async_ld16(unsigned lds_off, const void* g)
{
    asm volatile("global_load_async_to_lds_b128 %0, %1, off"
                 :: "v"(lds_off), "v"((unsigned long long)(uintptr_t)g)
                 : "memory");
}
__device__ __forceinline__ void wait_async0()
{
    asm volatile("s_wait_asynccnt 0x0" ::: "memory");
}

// ---- CDNA5 Tensor Data Mover: 2-D tile, global -> LDS (TENSORcnt) ---------
// D# group0/group1 packed per ISA 8.3/8.4: count=1, type=2 ("image"),
// data_size=2 bytes, tile (tile_d0 x tile_d1), row stride stride0 elems.
__device__ __forceinline__ void tdm_load_2d(unsigned lds_addr, const void* gptr,
                                            unsigned tile_d0, unsigned tile_d1,
                                            unsigned tens_d0, unsigned tens_d1,
                                            unsigned stride0)
{
    const unsigned long long ga = (unsigned long long)(uintptr_t)gptr;
    u32x4 g0;
    g0[0] = 1u;                                       // count=1 (valid), user
    g0[1] = lds_addr;                                 // LDS byte address
    g0[2] = (unsigned)ga;                             // global_addr[31:0]
    g0[3] = (unsigned)((ga >> 32) & 0x01FFFFFFu)      // global_addr[56:32]
          | (2u << 30);                               // type=2
    i32x8 g1;
    g1[0] = (int)(1u << 16);                          // data_size=1 -> 2 bytes
    g1[1] = (int)((tens_d0 & 0xFFFFu) << 16);         // tensor_dim0 lo16
    g1[2] = (int)(((tens_d0 >> 16) & 0xFFFFu) |       // tensor_dim0 hi16
                  ((tens_d1 & 0xFFFFu) << 16));       // tensor_dim1 lo16
    g1[3] = (int)(((tens_d1 >> 16) & 0xFFFFu) |       // tensor_dim1 hi16
                  ((tile_d0 & 0xFFFFu) << 16));       // tile_dim0
    g1[4] = (int)(tile_d1 & 0xFFFFu);                 // tile_dim1, tile_dim2=0
    g1[5] = (int)stride0;                             // tensor_dim0_stride lo32
    g1[6] = 0;                                        // stride hi, dim1_stride lo
    g1[7] = 0;                                        // dim1_stride hi
    asm volatile("tensor_load_to_lds %0, %1"          // VADDR2/3 = NULL (2-D)
                 :: "s"(g0), "s"(g1)
                 : "memory");
}
__device__ __forceinline__ void wait_tensor0()
{
    __builtin_amdgcn_s_wait_tensorcnt((unsigned short)0);
}

// ---- LDS fragment loads (contiguous k = 16h + s for both A and B) -------------------
__device__ __forceinline__ v16h ld_frag_a(const _Float16* base, int row, int ka)
{
    FragU u;
    u.u[0] = *(const uint4*)(base + row * B_TK + ka);
    u.u[1] = *(const uint4*)(base + row * B_TK + ka + 8);
    return u.v;
}
__device__ __forceinline__ v16h ld_frag_b(const _Float16* base, int col, int kb)
{
    FragU u;
    const uint4* p = (const uint4*)(base + col * B_TK + kb);
    u.u[0] = p[0];
    u.u[1] = p[1];
    return u.v;
}

// ---- fp32 -> f16 conversion pre-pass --------------------------------------
__global__ __launch_bounds__(256) void k_cvt(const float* __restrict__ in,
                                             _Float16* __restrict__ outp, int n4)
{
    const int i = blockIdx.x * blockDim.x + threadIdx.x;
    if (i < n4) {
        const float4 f = ((const float4*)in)[i];
        H4 o;
        o.h[0] = (_Float16)f.x; o.h[1] = (_Float16)f.y;
        o.h[2] = (_Float16)f.z; o.h[3] = (_Float16)f.w;
        ((uint2*)outp)[i] = o.u;
    }
}

// W2 [1000,1024] fp32 -> [1024,1024] f16, zero-padded rows >= 1000
__global__ __launch_bounds__(256) void k_cvt_pad(const float* __restrict__ W2,
                                                 _Float16* __restrict__ outp)
{
    const int i   = blockIdx.x * blockDim.x + threadIdx.x;  // vec4 index
    const int row = (i * 4) >> 10;
    H4 o;
    if (row < NCLS) {
        const float4 f = ((const float4*)W2)[i];
        o.h[0] = (_Float16)f.x; o.h[1] = (_Float16)f.y;
        o.h[2] = (_Float16)f.z; o.h[3] = (_Float16)f.w;
    } else {
        o.h[0] = o.h[1] = o.h[2] = o.h[3] = (_Float16)0.0f;
    }
    ((uint2*)outp)[i] = o.u;
}

// ---------------- GEMM1: h[m,n] = sum_k xh[m,k]*w1h[n,k] + b1[n] -----------
__global__ __launch_bounds__(128) void k_gemm1(const _Float16* __restrict__ xh,
                                               const _Float16* __restrict__ w1h,
                                               const float* __restrict__ b1,
                                               float* __restrict__ h)
{
    __shared__ __align__(16) _Float16 As[2][B_TM][B_TK];   // 2 x 4 KiB
    __shared__ __align__(16) _Float16 Bs[2][B_TN][B_TK];   // 2 x 8 KiB

    const int m0   = blockIdx.x * B_TM;
    const int n0   = blockIdx.y * B_TN;
    const int tid  = threadIdx.x;
    const int wave = tid >> 5, lane = tid & 31;
    const int wn   = wave * 32;
    const int l16  = lane & 15;
    const bool hi  = (lane >= 16);
    // wave-uniform scalar (SGPR) -> guaranteed scalar branch for TDM issue
    const bool w0  = (__builtin_amdgcn_readfirstlane(wave) == 0);

    const unsigned ldsA[2] = { (unsigned)(uintptr_t)&As[0][0][0],
                               (unsigned)(uintptr_t)&As[1][0][0] };
    const unsigned ldsB[2] = { (unsigned)(uintptr_t)&Bs[0][0][0],
                               (unsigned)(uintptr_t)&Bs[1][0][0] };

#if USE_TDM
    auto issue = [&](int buf, int k0) {
        if (w0) {
            tdm_load_2d(ldsA[buf], xh  + (size_t)m0 * DIM + k0,
                        B_TK, B_TM, DIM, M1,  DIM);
            tdm_load_2d(ldsB[buf], w1h + (size_t)n0 * DIM + k0,
                        B_TK, B_TN, DIM, DIM, DIM);
        }
    };
#else
    auto issue = [&](int buf, int k0) {
        #pragma unroll
        for (int j = 0; j < 2; ++j) {          // A: 256 16B chunks
            const int ch = tid + j * 128;
            const int r = ch >> 1, c8 = (ch & 1) * 8;
            async_ld16(ldsA[buf] + (unsigned)(r * B_TK + c8) * 2,
                       xh + (size_t)(m0 + r) * DIM + k0 + c8);
        }
        #pragma unroll
        for (int j = 0; j < 4; ++j) {          // B: 512 16B chunks
            const int ch = tid + j * 128;
            const int r = ch >> 1, c8 = (ch & 1) * 8;
            async_ld16(ldsB[buf] + (unsigned)(r * B_TK + c8) * 2,
                       w1h + (size_t)(n0 + r) * DIM + k0 + c8);
        }
    };
#endif

    v8f acc[4][2] = {};
    issue(0, 0);
    int cur = 0;
    for (int k0 = 0; k0 < DIM; k0 += B_TK) {
#if USE_TDM
        wait_tensor0();                        // no-op for waves with cnt==0
#else
        wait_async0();
#endif
        __syncthreads();                       // tile[cur] visible to all
        if (k0 + B_TK < DIM) issue(cur ^ 1, k0 + B_TK);

        const _Float16* Ab = &As[cur][0][0];
        const _Float16* Bb = &Bs[cur][0][0];
        v16h bf0 = ld_frag_b(Bb, wn + l16,      hi ? 16 : 0);
        v16h bf1 = ld_frag_b(Bb, wn + 16 + l16, hi ? 16 : 0);
        #pragma unroll
        for (int mt = 0; mt < 4; ++mt) {
            const v16h af = ld_frag_a(Ab, mt * 16 + l16, hi ? 16 : 0);
            acc[mt][0] = __builtin_amdgcn_wmma_f32_16x16x32_f16(
                false, af, false, bf0, (short)0, acc[mt][0], false, false);
            acc[mt][1] = __builtin_amdgcn_wmma_f32_16x16x32_f16(
                false, af, false, bf1, (short)0, acc[mt][1], false, false);
        }
        __syncthreads();                       // done reading tile[cur]
        cur ^= 1;
    }

    #pragma unroll
    for (int nt = 0; nt < 2; ++nt) {
        const int   nc   = n0 + wn + nt * 16 + l16;
        const float bias = b1[nc];
        #pragma unroll
        for (int mt = 0; mt < 4; ++mt)
            #pragma unroll
            for (int r = 0; r < 8; ++r)
                h[(size_t)(m0 + mt * 16 + r + (hi ? 8 : 0)) * DIM + nc] =
                    acc[mt][nt][r] + bias;
    }
}

// -------- LIF scan: v += (h-v)/2; spike; hard reset (2 elems/thread) -------
__global__ __launch_bounds__(256) void k_lif(const float* __restrict__ h,
                                             _Float16* __restrict__ s)
{
    const int i = blockIdx.x * blockDim.x + threadIdx.x;   // pair index
    float v0 = 0.0f, v1 = 0.0f;
    #pragma unroll
    for (int t = 0; t < T_STEPS; ++t) {
        const float2 ht = ((const float2*)(h + (size_t)t * (BATCH * DIM)))[i];
        v0 = v0 + (ht.x - v0) * 0.5f;
        v1 = v1 + (ht.y - v1) * 0.5f;
        const float s0 = (v0 >= 1.0f) ? 1.0f : 0.0f;
        const float s1 = (v1 >= 1.0f) ? 1.0f : 0.0f;
        v0 *= (1.0f - s0);
        v1 *= (1.0f - s1);
        H2 o;
        o.h[0] = (_Float16)s0;
        o.h[1] = (_Float16)s1;
        ((unsigned*)(s + (size_t)t * (BATCH * DIM)))[i] = o.u;
    }
}

// ------ GEMM2 split-K over t: ypart[z][b][c] = sum_{t in z,d} s*W2 ---------
__global__ __launch_bounds__(128) void k_gemm2(const _Float16* __restrict__ s,
                                               const _Float16* __restrict__ w2h,
                                               float* __restrict__ ypart)
{
    __shared__ __align__(16) _Float16 As[2][B_TM][B_TK];
    __shared__ __align__(16) _Float16 Bs[2][B_TN][B_TK];

    const int b0   = blockIdx.x * B_TM;     // batch tile
    const int c0   = blockIdx.y * B_TN;     // class tile (padded to 1024)
    const int z    = blockIdx.z;            // t in [4z, 4z+4)
    const int tid  = threadIdx.x;
    const int wave = tid >> 5, lane = tid & 31;
    const int wn   = wave * 32;
    const int l16  = lane & 15;
    const bool hi  = (lane >= 16);
    const bool w0  = (__builtin_amdgcn_readfirstlane(wave) == 0);

    const unsigned ldsA[2] = { (unsigned)(uintptr_t)&As[0][0][0],
                               (unsigned)(uintptr_t)&As[1][0][0] };
    const unsigned ldsB[2] = { (unsigned)(uintptr_t)&Bs[0][0][0],
                               (unsigned)(uintptr_t)&Bs[1][0][0] };

#if USE_TDM
    auto issue = [&](int buf, int kk) {
        if (w0) {
            const int t  = z * 4 + (kk >> 10);
            const int k0 = kk & (DIM - 1);
            tdm_load_2d(ldsA[buf], s + ((size_t)t * BATCH + b0) * DIM + k0,
                        B_TK, B_TM, DIM, M1,  DIM);
            tdm_load_2d(ldsB[buf], w2h + (size_t)c0 * DIM + k0,
                        B_TK, B_TN, DIM, DIM, DIM);
        }
    };
#else
    auto issue = [&](int buf, int kk) {
        const int t  = z * 4 + (kk >> 10);
        const int k0 = kk & (DIM - 1);
        #pragma unroll
        for (int j = 0; j < 2; ++j) {
            const int ch = tid + j * 128;
            const int r = ch >> 1, c8 = (ch & 1) * 8;
            async_ld16(ldsA[buf] + (unsigned)(r * B_TK + c8) * 2,
                       s + ((size_t)t * BATCH + b0 + r) * DIM + k0 + c8);
        }
        #pragma unroll
        for (int j = 0; j < 4; ++j) {
            const int ch = tid + j * 128;
            const int r = ch >> 1, c8 = (ch & 1) * 8;
            async_ld16(ldsB[buf] + (unsigned)(r * B_TK + c8) * 2,
                       w2h + (size_t)(c0 + r) * DIM + k0 + c8);
        }
    };
#endif

    v8f acc[4][2] = {};
    issue(0, 0);
    int cur = 0;
    const int KTOT = 4 * DIM;
    for (int kk = 0; kk < KTOT; kk += B_TK) {
#if USE_TDM
        wait_tensor0();
#else
        wait_async0();
#endif
        __syncthreads();
        if (kk + B_TK < KTOT) issue(cur ^ 1, kk + B_TK);

        const _Float16* Ab = &As[cur][0][0];
        const _Float16* Bb = &Bs[cur][0][0];
        v16h bf0 = ld_frag_b(Bb, wn + l16,      hi ? 16 : 0);
        v16h bf1 = ld_frag_b(Bb, wn + 16 + l16, hi ? 16 : 0);
        #pragma unroll
        for (int mt = 0; mt < 4; ++mt) {
            const v16h af = ld_frag_a(Ab, mt * 16 + l16, hi ? 16 : 0);
            acc[mt][0] = __builtin_amdgcn_wmma_f32_16x16x32_f16(
                false, af, false, bf0, (short)0, acc[mt][0], false, false);
            acc[mt][1] = __builtin_amdgcn_wmma_f32_16x16x32_f16(
                false, af, false, bf1, (short)0, acc[mt][1], false, false);
        }
        __syncthreads();
        cur ^= 1;
    }

    #pragma unroll
    for (int nt = 0; nt < 2; ++nt) {
        const int cc = c0 + wn + nt * 16 + l16;
        #pragma unroll
        for (int mt = 0; mt < 4; ++mt)
            #pragma unroll
            for (int r = 0; r < 8; ++r) {
                const int bb = b0 + mt * 16 + r + (hi ? 8 : 0);
                ypart[((size_t)z * BATCH + bb) * 1024 + cc] = acc[mt][nt][r];
            }
    }
}

// ------- reduce split-K, apply 1/T and bias, log_softmax over C ------------
__global__ __launch_bounds__(128) void k_softmax(const float* __restrict__ ypart,
                                                 const float* __restrict__ b2,
                                                 float* __restrict__ out)
{
    __shared__ float red[128];
    const int b   = blockIdx.x;
    const int tid = threadIdx.x;

    float yv[8];
    float lmax = -3.4e38f;
    #pragma unroll
    for (int j = 0; j < 8; ++j) {
        const int c = tid + j * 128;
        float a = 0.0f;
        if (c < NCLS) {
            #pragma unroll
            for (int z = 0; z < 8; ++z)
                a += ypart[((size_t)z * BATCH + b) * 1024 + c];
            a = a * (1.0f / (float)T_STEPS) + b2[c];
            lmax = fmaxf(lmax, a);
        }
        yv[j] = a;
    }

    red[tid] = lmax;
    __syncthreads();
    for (int sft = 64; sft > 0; sft >>= 1) {
        if (tid < sft) red[tid] = fmaxf(red[tid], red[tid + sft]);
        __syncthreads();
    }
    const float mx = red[0];
    __syncthreads();

    float lsum = 0.0f;
    #pragma unroll
    for (int j = 0; j < 8; ++j) {
        const int c = tid + j * 128;
        if (c < NCLS) lsum += __expf(yv[j] - mx);
    }
    red[tid] = lsum;
    __syncthreads();
    for (int sft = 64; sft > 0; sft >>= 1) {
        if (tid < sft) red[tid] += red[tid + sft];
        __syncthreads();
    }
    const float lse = __logf(red[0]);

    #pragma unroll
    for (int j = 0; j < 8; ++j) {
        const int c = tid + j * 128;
        if (c < NCLS) out[(size_t)b * NCLS + c] = (yv[j] - mx) - lse;
    }
}

// ---------------------------------------------------------------------------
extern "C" void kernel_launch(void* const* d_in, const int* in_sizes, int n_in,
                              void* d_out, int out_size, void* d_ws, size_t ws_size,
                              hipStream_t stream)
{
    (void)in_sizes; (void)n_in; (void)out_size; (void)ws_size;

    const float* x  = (const float*)d_in[0];   // [T,B,D]
    const float* W1 = (const float*)d_in[1];   // [D,D]
    const float* b1 = (const float*)d_in[2];   // [D]
    const float* W2 = (const float*)d_in[3];   // [C,D]
    const float* b2 = (const float*)d_in[4];   // [C]
    float*       out = (float*)d_out;          // [B,C]

    char* ws = (char*)d_ws;
    float*    h     = (float*)(ws);                         // 64 MiB
    _Float16* s     = (_Float16*)(ws + (size_t)67108864);   // 32 MiB
    float*    ypart = (float*)(ws + (size_t)100663296);     // 16 MiB
    _Float16* xh    = (_Float16*)(ws + (size_t)117440512);  // 32 MiB
    _Float16* w1h   = (_Float16*)(ws + (size_t)150994944);  //  2 MiB
    _Float16* w2h   = (_Float16*)(ws + (size_t)153092096);  //  2 MiB (padded)

    // fp32 -> f16 pre-pass
    k_cvt    <<<dim3(16384), 256, 0, stream>>>(x,  xh,  (M1 * DIM) / 4);
    k_cvt    <<<dim3(1024),  256, 0, stream>>>(W1, w1h, (DIM * DIM) / 4);
    k_cvt_pad<<<dim3(1024),  256, 0, stream>>>(W2, w2h);

    // GEMM1: M=16384, N=1024 -> 256 x 8 tiles of 64x128
    k_gemm1<<<dim3(M1 / B_TM, DIM / B_TN, 1), 128, 0, stream>>>(xh, w1h, b1, h);

    // LIF: B*D/2 = 262144 pairs
    k_lif<<<dim3((BATCH * DIM) / 512), 256, 0, stream>>>(h, s);

    // GEMM2: M=512, Npad=1024, split-K over 8 t-groups
    k_gemm2<<<dim3(BATCH / B_TM, 1024 / B_TN, 8), 128, 0, stream>>>(s, w2h, ypart);

    // reduce + bias + log_softmax
    k_softmax<<<dim3(BATCH), 128, 0, stream>>>(ypart, b2, out);
}
